// VarianceAdaptor_50182397886799
// MI455X (gfx1250) — matrix operand order repaired and checked by reading, not compiled, not run on hardware
//
#include <hip/hip_runtime.h>
#include <hip/hip_bf16.h>
#include <stdint.h>

// ---------------------------------------------------------------------------
// Shapes (fixed by the reference)
// ---------------------------------------------------------------------------
#define BB 64
#define TT 1024
#define DD 256
#define FF 256
#define KK 3
#define MAXLEN 8192

typedef __attribute__((ext_vector_type(16))) __bf16 v16bf;
typedef __attribute__((ext_vector_type(8)))  float  v8f;

// ---------------------------------------------------------------------------
// Weight prep: w (K,D,F) f32 -> wT [k][f][d] bf16 (so B-matrix K-dim (=d) is
// contiguous; a WMMA B-fragment becomes one 32-byte contiguous load)
// ---------------------------------------------------------------------------
__global__ __launch_bounds__(256)
void wprep_kernel(const float* __restrict__ w, __bf16* __restrict__ wT) {
    int idx = blockIdx.x * 256 + threadIdx.x;      // < 3*256*256
    int kk  = idx >> 16;
    int rem = idx & 65535;
    int d   = rem >> 8;
    int f   = rem & 255;
    wT[((size_t)kk * 256 + f) * 256 + d] = (__bf16)w[idx];
}

// ---------------------------------------------------------------------------
// Prep: searchsorted (side=left) for pitch/energy bins, x->bf16, (x+pemb)->bf16
// One block per (b,t) row; 256 threads over d.
// ---------------------------------------------------------------------------
__global__ __launch_bounds__(256)
void prep_kernel(const float* __restrict__ x,
                 const float* __restrict__ pitch_t,
                 const float* __restrict__ energy_t,
                 const float* __restrict__ pitch_bins,
                 const float* __restrict__ energy_bins,
                 const float* __restrict__ pitch_emb,
                 __bf16* __restrict__ xbf,
                 __bf16* __restrict__ x2bf,
                 int* __restrict__ pidx,
                 int* __restrict__ eidx) {
    int bt = blockIdx.x;
    int d  = threadIdx.x;
    float pv = pitch_t[bt];
    float ev = energy_t[bt];
    int lo = 0, hi = 255;                          // lower_bound over 255 bins
    while (lo < hi) { int m = (lo + hi) >> 1; if (pitch_bins[m] < pv) lo = m + 1; else hi = m; }
    int pi = lo;
    lo = 0; hi = 255;
    while (lo < hi) { int m = (lo + hi) >> 1; if (energy_bins[m] < ev) lo = m + 1; else hi = m; }
    int ei = lo;
    if (d == 0) { pidx[bt] = pi; eidx[bt] = ei; }
    size_t o = (size_t)bt * DD + d;
    float xv = x[o];
    xbf[o]  = (__bf16)xv;
    x2bf[o] = (__bf16)(xv + pitch_emb[(size_t)pi * DD + d]);
}

// ---------------------------------------------------------------------------
// Per-batch cumsum of max(dur,0); also emits duration (as f32) and mel_len.
// ---------------------------------------------------------------------------
__global__ __launch_bounds__(256)
void cumsum_kernel(const int* __restrict__ dur,
                   int* __restrict__ cum,
                   float* __restrict__ dur_out,
                   float* __restrict__ mellen_out) {
    int b = blockIdx.x, tid = threadIdx.x;
    __shared__ int part[256];
    int v[4]; int s = 0;
    for (int i = 0; i < 4; ++i) {
        int e = tid * 4 + i;
        int dv0 = dur[(size_t)b * TT + e];
        dur_out[(size_t)b * TT + e] = (float)dv0;
        int dv = dv0 < 0 ? 0 : dv0;
        s += dv; v[i] = s;
    }
    part[tid] = s;
    __syncthreads();
    for (int off = 1; off < 256; off <<= 1) {
        int add = (tid >= off) ? part[tid - off] : 0;
        __syncthreads();
        part[tid] += add;
        __syncthreads();
    }
    int excl = part[tid] - s;
    for (int i = 0; i < 4; ++i)
        cum[(size_t)b * TT + tid * 4 + i] = excl + v[i];
    if (tid == 255) mellen_out[b] = (float)(excl + s);
}

// ---------------------------------------------------------------------------
// Fused conv1d(K=3, 256->256, pad=1) + bias + ReLU + LayerNorm (over F).
// One block = 16 consecutive t-rows x full F=256, 8 waves x 2 N-tiles each.
// K-reduction = 3 taps * 256 ch = 24 WMMA steps of 32 (bf16, f32 accum).
// FUSE variant additionally applies the F->1 linear + source mask.
// ---------------------------------------------------------------------------
template <bool FUSE>
__global__ __launch_bounds__(256)
void convln_kernel(const __bf16* __restrict__ act,    // (B,T,256) bf16
                   const __bf16* __restrict__ wT,     // (3,256,256) [k][f][d] bf16
                   const float* __restrict__ bias,
                   const float* __restrict__ gamma,
                   const float* __restrict__ beta,
                   const float* __restrict__ lw,      // FUSE only
                   const float* __restrict__ lb,      // FUSE only
                   const unsigned char* __restrict__ mask, // FUSE only (B,T) bool
                   __bf16* __restrict__ outact,       // !FUSE (B,T,256) bf16
                   float* __restrict__ outpred) {     // FUSE (B,T) f32
    const int blk = blockIdx.x;
    const int b   = blk >> 6;            // T/16 = 64 tiles per batch
    const int t0  = (blk & 63) << 4;
    const int tid = threadIdx.x;
    const int wave = tid >> 5;
    const int lane = tid & 31;

    __shared__ __bf16 As[18 * 264];      // rows t0-1 .. t0+16 (conv halo)
    __shared__ float  Hs[16 * 264];      // relu(conv+bias)
    __shared__ float2 red[16][16];
    __shared__ float2 stats[16];

    for (int idx = tid; idx < 18 * 256; idx += 256) {
        int r = idx >> 8, d = idx & 255;
        int t = t0 + r - 1;
        __bf16 v = (__bf16)0.0f;
        if (t >= 0 && t < TT) v = act[((size_t)b * TT + t) * DD + d];
        As[r * 264 + d] = v;
    }
    __syncthreads();

    // WMMA register-layout parameters (wave32, 16-bit A / B, f32 C/D)
    const int mrow = lane & 15;                  // A row
    const int kbA  = (lane < 16) ? 0 : 16;       // A k-halves
    const int kbB  = (lane < 16) ? 0 : 16;       // B k-halves
    const int f0   = wave * 32 + (lane & 15);    // N-tile 0 column
    const int f1   = f0 + 16;                    // N-tile 1 column

    v8f acc0 = {}; v8f acc1 = {};
    for (int s = 0; s < 24; ++s) {
        const int kk = s >> 3;
        const int d0 = (s & 7) << 5;
        // A fragment from LDS (row = t-row + tap)
        v16bf a;
        const int rowoff = (mrow + kk) * 264 + d0;
#pragma unroll
        for (int j = 0; j < 8; ++j) {
            const int koff = kbA + 2 * j;
            a[2 * j]     = As[rowoff + koff];
            a[2 * j + 1] = As[rowoff + koff + 1];
        }
        // B fragments: contiguous 32B loads thanks to [k][f][d] layout
        const __bf16* bp0 = wT + ((size_t)(kk * 256 + f0) * 256 + d0 + kbB);
        const __bf16* bp1 = wT + ((size_t)(kk * 256 + f1) * 256 + d0 + kbB);
        v16bf bm0 = *(const v16bf*)bp0;
        v16bf bm1 = *(const v16bf*)bp1;
        acc0 = __builtin_amdgcn_wmma_f32_16x16x32_bf16(false, a, false, bm0,
                                                       (short)0, acc0, false, false);
        acc1 = __builtin_amdgcn_wmma_f32_16x16x32_bf16(false, a, false, bm1,
                                                       (short)0, acc1, false, false);
    }

    // bias + relu -> LDS (C/D layout: VGPR r -> row r (+8 for hi lanes))
    const float bia0 = bias[f0];
    const float bia1 = bias[f1];
    const int rbase = (lane < 16) ? 0 : 8;
#pragma unroll
    for (int r = 0; r < 8; ++r) {
        float v0 = acc0[r] + bia0; if (v0 < 0.f) v0 = 0.f;
        float v1 = acc1[r] + bia1; if (v1 < 0.f) v1 = 0.f;
        Hs[(rbase + r) * 264 + f0] = v0;
        Hs[(rbase + r) * 264 + f1] = v1;
    }
    __syncthreads();

    // LayerNorm stats over F=256 per row (16 threads x 16 elems per row)
    {
        int row = tid >> 4, seg = tid & 15;
        float s = 0.f, sq = 0.f;
#pragma unroll
        for (int i = 0; i < 16; ++i) {
            float v = Hs[row * 264 + seg * 16 + i];
            s += v; sq += v * v;
        }
        red[row][seg] = make_float2(s, sq);
    }
    __syncthreads();
    if (tid < 16) {
        float s = 0.f, sq = 0.f;
        for (int i = 0; i < 16; ++i) { s += red[tid][i].x; sq += red[tid][i].y; }
        float m = s * (1.0f / 256.0f);
        float var = sq * (1.0f / 256.0f) - m * m;
        stats[tid] = make_float2(m, rsqrtf(var + 1e-5f));
    }
    __syncthreads();

    if (!FUSE) {
        for (int idx = tid; idx < 16 * 256; idx += 256) {
            int row = idx >> 8, f = idx & 255;
            float2 st = stats[row];
            float v = (Hs[row * 264 + f] - st.x) * st.y * gamma[f] + beta[f];
            outact[((size_t)b * TT + t0 + row) * FF + f] = (__bf16)v;
        }
    } else {
        int row = tid >> 4, seg = tid & 15;
        float2 st = stats[row];
        float s = 0.f;
        for (int i = 0; i < 16; ++i) {
            int f = seg * 16 + i;
            float v = (Hs[row * 264 + f] - st.x) * st.y * gamma[f] + beta[f];
            s += v * lw[f];
        }
        red[row][seg].x = s;
        __syncthreads();
        if (tid < 16) {
            float s2 = 0.f;
            for (int i = 0; i < 16; ++i) s2 += red[tid][i].x;
            s2 += lb[0];
            size_t o = (size_t)b * TT + t0 + tid;
            if (mask[o]) s2 = 0.f;
            outpred[o] = s2;
        }
    }
}

// ---------------------------------------------------------------------------
// Length regulator: per (b,pos), searchsorted(cum, pos, right), gather row of
// x + pitch_emb + energy_emb (computed on the fly), mask past mel_len.
// Bandwidth-bound: 512 MB of writes -> the ~23us floor at 23.3 TB/s.
// ---------------------------------------------------------------------------
__global__ __launch_bounds__(256)
void gather_kernel(const float* __restrict__ x,
                   const float* __restrict__ pitch_emb,
                   const float* __restrict__ energy_emb,
                   const int* __restrict__ pidx,
                   const int* __restrict__ eidx,
                   const int* __restrict__ cum,
                   float* __restrict__ out,
                   float* __restrict__ mask_out) {
    int bp  = blockIdx.x;                 // b*MAXLEN + pos
    int b   = bp >> 13;
    int pos = bp & (MAXLEN - 1);
    int d   = threadIdx.x;
    const int* c = cum + (size_t)b * TT;
    int mel = c[TT - 1];
    int lo = 0, hi = TT;                  // upper_bound: first i with c[i] > pos
    while (lo < hi) { int m = (lo + hi) >> 1; if (c[m] <= pos) lo = m + 1; else hi = m; }
    int idx = lo > (TT - 1) ? (TT - 1) : lo;
    bool valid = pos < mel;
    float v = 0.f;
    if (valid) {
        size_t bt = (size_t)b * TT + idx;
        v = x[bt * DD + d]
          + pitch_emb[(size_t)pidx[bt] * DD + d]
          + energy_emb[(size_t)eidx[bt] * DD + d];
    }
    out[(size_t)bp * DD + d] = v;
    if (d == 0) mask_out[bp] = valid ? 0.f : 1.f;
}

// ---------------------------------------------------------------------------
// Host launcher
// ---------------------------------------------------------------------------
extern "C" void kernel_launch(void* const* d_in, const int* in_sizes, int n_in,
                              void* d_out, int out_size, void* d_ws, size_t ws_size,
                              hipStream_t stream) {
    (void)n_in; (void)out_size; (void)ws_size;
    const float* x            = (const float*)d_in[0];
    const unsigned char* smsk = (const unsigned char*)d_in[1];
    const float* pitch_t      = (const float*)d_in[2];
    const float* energy_t     = (const float*)d_in[3];
    const int*   dur_t        = (const int*)d_in[4];
    // d_in[5] = max_len scalar (fixed 8192 here)

    // --- adaptive param-leaf ordering (insertion order vs JAX sorted keys) ---
    const int base = 6;
    int I[10]; // w1 b1 g1 be1 w2 b2 g2 be2 lw lb  (our canonical order)
    if (in_sizes[base] == KK * DD * FF) {          // insertion order inside predictor
        int m[10] = {0, 1, 2, 3, 4, 5, 6, 7, 8, 9};
        for (int i = 0; i < 10; ++i) I[i] = m[i];
    } else {                                        // sorted: b1 b2 be1 be2 g1 g2 lb lw w1 w2
        int m[10] = {8, 0, 4, 2, 9, 1, 5, 3, 7, 6};
        for (int i = 0; i < 10; ++i) I[i] = m[i];
    }
    int durB, pitB, eneB, pembI, eembI, pbinI, ebinI;
    if (in_sizes[base + 20] != 255) {               // outer insertion order
        durB = 6; pitB = 16; eneB = 26; pembI = 36; eembI = 37; pbinI = 38; ebinI = 39;
    } else {                                        // outer sorted order
        durB = 6; eneB = 16; ebinI = 26; eembI = 27; pitB = 28; pbinI = 38; pembI = 39;
    }
    #define PW1(B)  ((const float*)d_in[(B) + I[0]])
    #define PB1(B)  ((const float*)d_in[(B) + I[1]])
    #define PG1(B)  ((const float*)d_in[(B) + I[2]])
    #define PBE1(B) ((const float*)d_in[(B) + I[3]])
    #define PW2(B)  ((const float*)d_in[(B) + I[4]])
    #define PB2(B)  ((const float*)d_in[(B) + I[5]])
    #define PG2(B)  ((const float*)d_in[(B) + I[6]])
    #define PBE2(B) ((const float*)d_in[(B) + I[7]])
    #define PLW(B)  ((const float*)d_in[(B) + I[8]])
    #define PLB(B)  ((const float*)d_in[(B) + I[9]])
    const float* pemb  = (const float*)d_in[pembI];
    const float* eemb  = (const float*)d_in[eembI];
    const float* pbins = (const float*)d_in[pbinI];
    const float* ebins = (const float*)d_in[ebinI];

    // --- workspace layout ---
    char* ws = (char*)d_ws;
    const size_t ACT = (size_t)BB * TT * DD;        // 16M elems
    __bf16* xbf  = (__bf16*)(ws);
    __bf16* x2bf = (__bf16*)(ws + ACT * 2);
    __bf16* h1bf = (__bf16*)(ws + ACT * 4);
    __bf16* wT0  = (__bf16*)(ws + ACT * 6);         // 6 x (3*256*256) bf16
    const size_t WSZ = (size_t)KK * DD * FF;
    __bf16* wTd1 = wT0;            __bf16* wTd2 = wT0 + 1 * WSZ;
    __bf16* wTp1 = wT0 + 2 * WSZ;  __bf16* wTp2 = wT0 + 3 * WSZ;
    __bf16* wTe1 = wT0 + 4 * WSZ;  __bf16* wTe2 = wT0 + 5 * WSZ;
    char* ws2 = ws + ACT * 6 + 6 * WSZ * 2 + 256;
    int* pidx = (int*)(ws2);
    int* eidx = (int*)(ws2 + (size_t)BB * TT * 4);
    int* cum  = (int*)(ws2 + (size_t)BB * TT * 8);

    // --- output layout (flat, reference tuple order) ---
    float* outF       = (float*)d_out;
    float* out_mel    = outF;                                   // B*MAXLEN*D
    float* out_pitch  = outF + (size_t)BB * MAXLEN * DD;
    float* out_energy = out_pitch + (size_t)BB * TT;
    float* out_logdur = out_energy + (size_t)BB * TT;
    float* out_dur    = out_logdur + (size_t)BB * TT;
    float* out_mellen = out_dur + (size_t)BB * TT;
    float* out_mask   = out_mellen + BB;

    const int wgrid = (KK * DD * FF) / 256;         // 768
    wprep_kernel<<<wgrid, 256, 0, stream>>>(PW1(durB), wTd1);
    wprep_kernel<<<wgrid, 256, 0, stream>>>(PW2(durB), wTd2);
    wprep_kernel<<<wgrid, 256, 0, stream>>>(PW1(pitB), wTp1);
    wprep_kernel<<<wgrid, 256, 0, stream>>>(PW2(pitB), wTp2);
    wprep_kernel<<<wgrid, 256, 0, stream>>>(PW1(eneB), wTe1);
    wprep_kernel<<<wgrid, 256, 0, stream>>>(PW2(eneB), wTe2);

    prep_kernel<<<BB * TT, 256, 0, stream>>>(x, pitch_t, energy_t, pbins, ebins,
                                             pemb, xbf, x2bf, pidx, eidx);
    cumsum_kernel<<<BB, 256, 0, stream>>>(dur_t, cum, out_dur, out_mellen);

    const int cgrid = BB * TT / 16;                 // 4096
    // duration predictor (on x)
    convln_kernel<false><<<cgrid, 256, 0, stream>>>(xbf, wTd1, PB1(durB), PG1(durB), PBE1(durB),
                                                    nullptr, nullptr, nullptr, h1bf, nullptr);
    convln_kernel<true><<<cgrid, 256, 0, stream>>>(h1bf, wTd2, PB2(durB), PG2(durB), PBE2(durB),
                                                   PLW(durB), PLB(durB), smsk, nullptr, out_logdur);
    // pitch predictor (on x)
    convln_kernel<false><<<cgrid, 256, 0, stream>>>(xbf, wTp1, PB1(pitB), PG1(pitB), PBE1(pitB),
                                                    nullptr, nullptr, nullptr, h1bf, nullptr);
    convln_kernel<true><<<cgrid, 256, 0, stream>>>(h1bf, wTp2, PB2(pitB), PG2(pitB), PBE2(pitB),
                                                   PLW(pitB), PLB(pitB), smsk, nullptr, out_pitch);
    // energy predictor (on x + pitch_emb)
    convln_kernel<false><<<cgrid, 256, 0, stream>>>(x2bf, wTe1, PB1(eneB), PG1(eneB), PBE1(eneB),
                                                    nullptr, nullptr, nullptr, h1bf, nullptr);
    convln_kernel<true><<<cgrid, 256, 0, stream>>>(h1bf, wTe2, PB2(eneB), PG2(eneB), PBE2(eneB),
                                                   PLW(eneB), PLB(eneB), smsk, nullptr, out_energy);

    gather_kernel<<<BB * MAXLEN, 256, 0, stream>>>(x, pemb, eemb, pidx, eidx, cum,
                                                   out_mel, out_mask);
}
